// AGCRN_14628658610857
// MI455X (gfx1250) — compile-verified, matrix operand order checked by reading, not run on hardware
//
#include <hip/hip_runtime.h>
#include <hip/hip_bf16.h>

typedef __bf16 bf16_t;
typedef __bf16 v8bf  __attribute__((ext_vector_type(8)));
typedef __bf16 v16bf __attribute__((ext_vector_type(16)));
typedef float  v8f   __attribute__((ext_vector_type(8)));

#define B_   64
#define L_   12
#define N_   1024
#define E_   16
#define H_   64
#define HOR_ 12

// ---------------- WMMA fragment helpers (wave32, ISA 7.12.2 layouts) ----------

// A-matrix 16x32 bf16 fragment from row-major memory (global or LDS), ld in elems.
// lanes 0-15: M=lane,   K {0..7} then {16..23}
// lanes16-31: M=lane-16, K {8..15} then {24..31}
// -> per lane: two contiguous 16-byte chunks => 2x b128 loads.
__device__ __forceinline__ v16bf frag_a(const bf16_t* p, int ld, int row0, int k0) {
  int lane = threadIdx.x & 31;
  int m  = lane & 15;
  int kb = (lane < 16) ? 0 : 8;
  const v8bf* q = (const v8bf*)(p + (size_t)(row0 + m) * ld + k0 + kb);
  v8bf lo = q[0];
  v8bf hi = q[2];   // +16 elements
  return __builtin_shufflevector(lo, hi, 0, 1, 2, 3, 4, 5, 6, 7,
                                 8, 9, 10, 11, 12, 13, 14, 15);
}

// B-matrix 32x16 fragment from a pre-swizzled tile: tile = 32 lanes x 16 bf16,
// lane-contiguous (512 elems). One 32B contiguous read per lane, coalesced.
__device__ __forceinline__ v16bf frag_b_sw(const bf16_t* tile) {
  int lane = threadIdx.x & 31;
  const v8bf* q = (const v8bf*)(tile + lane * 16);
  v8bf lo = q[0];
  v8bf hi = q[1];
  return __builtin_shufflevector(lo, hi, 0, 1, 2, 3, 4, 5, 6, 7,
                                 8, 9, 10, 11, 12, 13, 14, 15);
}

// Swizzle map: element (kl in 0..31, cl in 0..15) of a 32x16 B tile -> lane*16+i
__device__ __forceinline__ int swz_idx(int kl, int cl) {
  int half = (kl >> 3) & 1;          // 0: k in {0..7,16..23} -> lanes 0-15
  int lane = half * 16 + cl;
  int i = (kl & 7) + ((kl >> 4) << 3);  // +8 if kl >= 16
  return lane * 16 + i;
}

__device__ __forceinline__ v8f zero8() {
  v8f z;
#pragma unroll
  for (int i = 0; i < 8; ++i) z[i] = 0.0f;
  return z;
}

__device__ __forceinline__ float sigmoidf_(float x) { return 1.0f / (1.0f + __expf(-x)); }

// ---------------- supports: A = softmax(relu(emb @ emb^T)), bf16 row-major ----
__global__ void ag_supports_kernel(const float* __restrict__ emb, bf16_t* __restrict__ A) {
  __shared__ float ei[E_];
  __shared__ float red[256];
  int i = blockIdx.x;
  if (threadIdx.x < E_) ei[threadIdx.x] = emb[i * E_ + threadIdx.x];
  __syncthreads();
  float vals[4];
  float mx = -1e30f;
#pragma unroll
  for (int s = 0; s < 4; ++s) {
    int j = threadIdx.x + 256 * s;
    float d = 0.0f;
#pragma unroll
    for (int e = 0; e < E_; ++e) d += ei[e] * emb[j * E_ + e];
    d = fmaxf(d, 0.0f);
    vals[s] = d;
    mx = fmaxf(mx, d);
  }
  red[threadIdx.x] = mx; __syncthreads();
  for (int off = 128; off > 0; off >>= 1) {
    if (threadIdx.x < off) red[threadIdx.x] = fmaxf(red[threadIdx.x], red[threadIdx.x + off]);
    __syncthreads();
  }
  mx = red[0]; __syncthreads();
  float sum = 0.0f;
#pragma unroll
  for (int s = 0; s < 4; ++s) { vals[s] = __expf(vals[s] - mx); sum += vals[s]; }
  red[threadIdx.x] = sum; __syncthreads();
  for (int off = 128; off > 0; off >>= 1) {
    if (threadIdx.x < off) red[threadIdx.x] += red[threadIdx.x + off];
    __syncthreads();
  }
  float inv = 1.0f / red[0];
#pragma unroll
  for (int s = 0; s < 4; ++s) {
    int j = threadIdx.x + 256 * s;
    A[(size_t)i * N_ + j] = (bf16_t)(vals[s] * inv);
  }
}

// ------------- per-node weights, written in B-fragment tile order ------------
// PW[n] covers Kd x O (Kd = 2*Cip padded); tiles (ktile, ct) of 32x16, 512 elems,
// lane-major inside. Value: sum_e emb[n,e] * w[e,k,c,o], zero for pad rows.
__global__ void ag_pw_kernel(const float* __restrict__ emb, const float* __restrict__ w,
                             bf16_t* __restrict__ PW, int Ci, int Cip, int O) {
  int per = 2 * Cip * O;                 // elems per node (Kd*O)
  long long idx = (long long)blockIdx.x * blockDim.x + threadIdx.x;
  if (idx >= (long long)N_ * per) return;
  int n = (int)(idx / per);
  int s = (int)(idx - (long long)n * per);
  int tcols = O >> 4;
  int tile = s >> 9;                     // 512 elems per tile
  int r = s & 511;
  int lane = r >> 4;
  int i = r & 15;
  int ktile = tile / tcols;
  int ct = tile - ktile * tcols;
  int half = lane >> 4;                  // 0 -> kb 0, 1 -> kb 8
  int kl = half * 8 + (i & 7) + ((i >> 3) << 4);  // inverse of fragment map
  int kr = ktile * 32 + kl;              // padded K index
  int o = ct * 16 + (lane & 15);
  int k = (kr >= Cip) ? 1 : 0;
  int c = kr - k * Cip;
  float acc = 0.0f;
  if (c < Ci) {
#pragma unroll
    for (int e = 0; e < E_; ++e)
      acc += emb[n * E_ + e] * w[(((size_t)e * 2 + k) * Ci + c) * O + o];
  }
  PW[idx] = (bf16_t)acc;
}

__global__ void ag_pb_kernel(const float* __restrict__ emb, const float* __restrict__ b,
                             float* __restrict__ PB, int O) {
  int idx = blockIdx.x * blockDim.x + threadIdx.x;
  if (idx >= N_ * O) return;
  int n = idx / O, o = idx - n * O;
  float acc = 0.0f;
#pragma unroll
  for (int e = 0; e < E_; ++e) acc += emb[n * E_ + e] * b[e * O + o];
  PB[idx] = acc;
}

__global__ void ag_zero_kernel(float* __restrict__ p, int n) {
  int idx = blockIdx.x * blockDim.x + threadIdx.x;
  if (idx < n) p[idx] = 0.0f;
}

// ------------- build X: row-major (for node G staging) + swizzled (for GEMM B)
__global__ void ag_build_x_kernel(const float* __restrict__ hist, const bf16_t* __restrict__ hs0,
                                  const float* __restrict__ h, const float* __restrict__ zr,
                                  bf16_t* __restrict__ X, bf16_t* __restrict__ Xsw,
                                  int layer, int t, int Cin, int Cip, int cand) {
  int idx = blockIdx.x * blockDim.x + threadIdx.x;
  int total = N_ * B_ * Cip;
  if (idx >= total) return;
  int Mcols = B_ * Cip;
  int n = idx / Mcols;
  int rem = idx - n * Mcols;
  int b = rem / Cip;
  int c = rem - b * Cip;
  float v;
  if (c < Cin) {
    if (layer == 0) v = hist[((size_t)b * L_ + t) * N_ + n];            // C==1
    else            v = (float)hs0[(((size_t)b * L_ + t) * N_ + n) * H_ + c];
  } else if (c < Cin + H_) {
    int s = c - Cin;
    v = h[((size_t)b * N_ + n) * H_ + s];
    if (cand) v *= zr[((size_t)b * N_ + n) * 128 + s];                  // z = first half
  } else {
    v = 0.0f;                                                           // K pad
  }
  bf16_t bv = (bf16_t)v;
  int col = b * Cip + c;
  X[(size_t)n * Mcols + col] = bv;
  // swizzled copy: rows = node index (K side of graph GEMM), cols = (b,c)
  int ktile = n >> 5, kl = n & 31;
  int ct = col >> 4, cl = col & 15;
  Xsw[(((size_t)ktile * (Mcols >> 4) + ct) << 9) + swz_idx(kl, cl)] = bv;
}

// ------------- graph GEMM: Y = A(1024x1024) @ X(1024xMcols) ------------------
// 128 threads / 4 waves; wave computes 32x64 of the 128x64 block tile.
// No LDS: A frags 2x b128 from row-major A, B frags 2x b128 from swizzled Xsw.
__global__ void ag_graph_gemm_kernel(const bf16_t* __restrict__ A, const bf16_t* __restrict__ Xsw,
                                     bf16_t* __restrict__ Y, int Mcols) {
  int rowBase = blockIdx.y * 128 + (threadIdx.x >> 5) * 32;
  int colBase = blockIdx.x * 64;
  int lane = threadIdx.x & 31;
  int tcols = Mcols >> 4;
  v8f acc[2][4];
#pragma unroll
  for (int s = 0; s < 2; ++s)
#pragma unroll
    for (int j = 0; j < 4; ++j) acc[s][j] = zero8();

  for (int kt = 0; kt < N_; kt += 32) {
    v16bf a0 = frag_a(A, N_, rowBase, kt);
    v16bf a1 = frag_a(A, N_, rowBase + 16, kt);
    const bf16_t* tb = Xsw + ((((size_t)(kt >> 5)) * tcols + (colBase >> 4)) << 9);
#pragma unroll
    for (int j = 0; j < 4; ++j) {
      v16bf bfr = frag_b_sw(tb + (j << 9));
      acc[0][j] = __builtin_amdgcn_wmma_f32_16x16x32_bf16(false, a0, false, bfr, (short)0,
                                                          acc[0][j], false, false);
      acc[1][j] = __builtin_amdgcn_wmma_f32_16x16x32_bf16(false, a1, false, bfr, (short)0,
                                                          acc[1][j], false, false);
    }
  }
  int nc = lane & 15;
  int mb = (lane < 16) ? 0 : 8;
#pragma unroll
  for (int s = 0; s < 2; ++s)
#pragma unroll
    for (int j = 0; j < 4; ++j)
#pragma unroll
      for (int v = 0; v < 8; ++v)
        Y[(size_t)(rowBase + 16 * s + mb + v) * Mcols + colBase + 16 * j + nc] =
            (bf16_t)acc[s][j][v];
}

// ------------- per-node gate GEMM: [X_n;Y_n](64 x Kd) @ W_n -> sigmoid -> zr -
__global__ void ag_node_gate_kernel(const bf16_t* __restrict__ X, const bf16_t* __restrict__ Y,
                                    const bf16_t* __restrict__ W, const float* __restrict__ PB,
                                    float* __restrict__ zr, int Cip) {
  __shared__ __align__(16) bf16_t G[64 * 256];
  int n = blockIdx.x;
  int Kd = 2 * Cip;
  const bf16_t* Xr = X + (size_t)n * (B_ * Cip);
  const bf16_t* Yr = Y + (size_t)n * (B_ * Cip);
  int rowCh = Cip >> 3;                        // 8-elem chunks per source row
  int nCh = 64 * 2 * rowCh;
  for (int ch = threadIdx.x; ch < nCh; ch += 256) {
    int b = ch / (2 * rowCh);
    int cw = ch - b * 2 * rowCh;
    const v8bf* src = (cw < rowCh) ? (const v8bf*)(Xr + (size_t)b * Cip) + cw
                                   : (const v8bf*)(Yr + (size_t)b * Cip) + (cw - rowCh);
    *((v8bf*)(G + b * Kd) + cw) = *src;
  }
  __syncthreads();
  int wave = threadIdx.x >> 5;
  int lane = threadIdx.x & 31;
  int rs = wave & 3;                           // 16-row strip of B
  int ch2 = wave >> 2;                         // 64-col half of O=128
  int ktiles = Kd >> 5;
  const bf16_t* Wn = W + (((size_t)n * ktiles * 8) << 9);   // 8 col tiles (O=128)
  v8f acc[4];
#pragma unroll
  for (int j = 0; j < 4; ++j) acc[j] = zero8();
  for (int kt = 0; kt < Kd; kt += 32) {
    v16bf af = frag_a(G, Kd, 16 * rs, kt);
    const bf16_t* tb = Wn + ((((size_t)(kt >> 5)) * 8 + ch2 * 4) << 9);
#pragma unroll
    for (int j = 0; j < 4; ++j) {
      v16bf bfr = frag_b_sw(tb + (j << 9));
      acc[j] = __builtin_amdgcn_wmma_f32_16x16x32_bf16(false, af, false, bfr, (short)0, acc[j],
                                                       false, false);
    }
  }
  int nc = lane & 15;
  int mb = (lane < 16) ? 0 : 8;
#pragma unroll
  for (int j = 0; j < 4; ++j)
#pragma unroll
    for (int v = 0; v < 8; ++v) {
      int b = 16 * rs + mb + v;
      int o = 64 * ch2 + 16 * j + nc;
      float val = sigmoidf_(acc[j][v] + PB[n * 128 + o]);
      zr[((size_t)b * N_ + n) * 128 + o] = val;
    }
}

// ------------- per-node update GEMM -> tanh -> GRU h update ------------------
__global__ void ag_node_upd_kernel(const bf16_t* __restrict__ X, const bf16_t* __restrict__ Y,
                                   const bf16_t* __restrict__ W, const float* __restrict__ PB,
                                   const float* __restrict__ zr, float* __restrict__ h,
                                   bf16_t* __restrict__ hs0, int Cip, int t, int layer) {
  __shared__ __align__(16) bf16_t G[64 * 256];
  int n = blockIdx.x;
  int Kd = 2 * Cip;
  const bf16_t* Xr = X + (size_t)n * (B_ * Cip);
  const bf16_t* Yr = Y + (size_t)n * (B_ * Cip);
  int rowCh = Cip >> 3;
  int nCh = 64 * 2 * rowCh;
  for (int ch = threadIdx.x; ch < nCh; ch += 128) {
    int b = ch / (2 * rowCh);
    int cw = ch - b * 2 * rowCh;
    const v8bf* src = (cw < rowCh) ? (const v8bf*)(Xr + (size_t)b * Cip) + cw
                                   : (const v8bf*)(Yr + (size_t)b * Cip) + (cw - rowCh);
    *((v8bf*)(G + b * Kd) + cw) = *src;
  }
  __syncthreads();
  int wave = threadIdx.x >> 5;                 // 0..3 -> 16-row strip of B
  int lane = threadIdx.x & 31;
  int ktiles = Kd >> 5;
  const bf16_t* Wn = W + (((size_t)n * ktiles * 4) << 9);   // 4 col tiles (O=64)
  v8f acc[4];
#pragma unroll
  for (int j = 0; j < 4; ++j) acc[j] = zero8();
  for (int kt = 0; kt < Kd; kt += 32) {
    v16bf af = frag_a(G, Kd, 16 * wave, kt);
    const bf16_t* tb = Wn + ((((size_t)(kt >> 5)) * 4) << 9);
#pragma unroll
    for (int j = 0; j < 4; ++j) {
      v16bf bfr = frag_b_sw(tb + (j << 9));
      acc[j] = __builtin_amdgcn_wmma_f32_16x16x32_bf16(false, af, false, bfr, (short)0, acc[j],
                                                       false, false);
    }
  }
  int nc = lane & 15;
  int mb = (lane < 16) ? 0 : 8;
#pragma unroll
  for (int j = 0; j < 4; ++j)
#pragma unroll
    for (int v = 0; v < 8; ++v) {
      int b = 16 * wave + mb + v;
      int o = 16 * j + nc;
      float hc = tanhf(acc[j][v] + PB[n * H_ + o]);
      float r  = zr[((size_t)b * N_ + n) * 128 + H_ + o];   // r = second half
      size_t hi = ((size_t)b * N_ + n) * H_ + o;
      float hn = r * h[hi] + (1.0f - r) * hc;
      h[hi] = hn;
      if (layer == 0) hs0[(((size_t)b * L_ + t) * N_ + n) * H_ + o] = (bf16_t)hn;
    }
}

// ------------- output head: out[b,o,n] = h[b,n,:] . conv_w[o,:] + conv_b[o] --
__global__ void ag_out_kernel(const float* __restrict__ h, const float* __restrict__ cw,
                              const float* __restrict__ cb, float* __restrict__ out) {
  int idx = blockIdx.x * blockDim.x + threadIdx.x;
  if (idx >= B_ * HOR_ * N_) return;
  int b = idx / (HOR_ * N_);
  int rem = idx - b * (HOR_ * N_);
  int o = rem / N_;
  int n = rem - o * N_;
  float acc = cb[o];
#pragma unroll
  for (int k = 0; k < H_; ++k) acc += h[((size_t)b * N_ + n) * H_ + k] * cw[o * H_ + k];
  out[idx] = acc;
}

// ----------------------------------------------------------------------------
extern "C" void kernel_launch(void* const* d_in, const int* in_sizes, int n_in,
                              void* d_out, int out_size, void* d_ws, size_t ws_size,
                              hipStream_t stream) {
  const float* hist = (const float*)d_in[0];
  const float* emb  = (const float*)d_in[1];
  const float* gw0  = (const float*)d_in[2];
  const float* gb0  = (const float*)d_in[3];
  const float* uw0  = (const float*)d_in[4];
  const float* ub0  = (const float*)d_in[5];
  const float* gw1  = (const float*)d_in[6];
  const float* gb1  = (const float*)d_in[7];
  const float* uw1  = (const float*)d_in[8];
  const float* ub1  = (const float*)d_in[9];
  const float* cw   = (const float*)d_in[10];
  const float* cb   = (const float*)d_in[11];
  float* out = (float*)d_out;

  char* p = (char*)d_ws;
  auto alloc = [&](size_t bytes) {
    char* r = p;
    p += (bytes + 255) & ~(size_t)255;
    return (void*)r;
  };
  bf16_t* A    = (bf16_t*)alloc((size_t)N_ * N_ * 2);
  bf16_t* PWg0 = (bf16_t*)alloc((size_t)N_ * 192 * 128 * 2);
  bf16_t* PWu0 = (bf16_t*)alloc((size_t)N_ * 192 * 64 * 2);
  bf16_t* PWg1 = (bf16_t*)alloc((size_t)N_ * 256 * 128 * 2);
  bf16_t* PWu1 = (bf16_t*)alloc((size_t)N_ * 256 * 64 * 2);
  float*  PBg0 = (float*)alloc((size_t)N_ * 128 * 4);
  float*  PBu0 = (float*)alloc((size_t)N_ * 64 * 4);
  float*  PBg1 = (float*)alloc((size_t)N_ * 128 * 4);
  float*  PBu1 = (float*)alloc((size_t)N_ * 64 * 4);
  bf16_t* X    = (bf16_t*)alloc((size_t)N_ * 8192 * 2);
  bf16_t* Xsw  = (bf16_t*)alloc((size_t)N_ * 8192 * 2);
  bf16_t* Y    = (bf16_t*)alloc((size_t)N_ * 8192 * 2);
  float*  zr   = (float*)alloc((size_t)B_ * N_ * 128 * 4);
  float*  h    = (float*)alloc((size_t)B_ * N_ * H_ * 4);
  bf16_t* hs0  = (bf16_t*)alloc((size_t)B_ * L_ * N_ * H_ * 2);

  // 1. supports
  ag_supports_kernel<<<N_, 256, 0, stream>>>(emb, A);

  // 2. per-node weights (bf16, fragment-swizzled, K padded 65->96) and biases
  {
    long long tot;
    tot = (long long)N_ * 2 * 96 * 128;
    ag_pw_kernel<<<(int)((tot + 255) / 256), 256, 0, stream>>>(emb, gw0, PWg0, 65, 96, 128);
    tot = (long long)N_ * 2 * 96 * 64;
    ag_pw_kernel<<<(int)((tot + 255) / 256), 256, 0, stream>>>(emb, uw0, PWu0, 65, 96, 64);
    tot = (long long)N_ * 2 * 128 * 128;
    ag_pw_kernel<<<(int)((tot + 255) / 256), 256, 0, stream>>>(emb, gw1, PWg1, 128, 128, 128);
    tot = (long long)N_ * 2 * 128 * 64;
    ag_pw_kernel<<<(int)((tot + 255) / 256), 256, 0, stream>>>(emb, uw1, PWu1, 128, 128, 64);
    int t2 = N_ * 128;
    ag_pb_kernel<<<(t2 + 255) / 256, 256, 0, stream>>>(emb, gb0, PBg0, 128);
    ag_pb_kernel<<<(t2 + 255) / 256, 256, 0, stream>>>(emb, gb1, PBg1, 128);
    t2 = N_ * 64;
    ag_pb_kernel<<<(t2 + 255) / 256, 256, 0, stream>>>(emb, ub0, PBu0, 64);
    ag_pb_kernel<<<(t2 + 255) / 256, 256, 0, stream>>>(emb, ub1, PBu1, 64);
  }

  const int hElems = B_ * N_ * H_;

  // 3. layer 0: Cin=1, Cip=96, Mcols = 6144
  ag_zero_kernel<<<(hElems + 255) / 256, 256, 0, stream>>>(h, hElems);
  {
    const int Cip = 96, Cin = 1, Mcols = B_ * Cip;
    const int xTot = N_ * B_ * Cip;
    for (int t = 0; t < L_; ++t) {
      ag_build_x_kernel<<<(xTot + 255) / 256, 256, 0, stream>>>(hist, hs0, h, zr, X, Xsw, 0, t, Cin, Cip, 0);
      ag_graph_gemm_kernel<<<dim3(Mcols / 64, N_ / 128), 128, 0, stream>>>(A, Xsw, Y, Mcols);
      ag_node_gate_kernel<<<N_, 256, 0, stream>>>(X, Y, PWg0, PBg0, zr, Cip);
      ag_build_x_kernel<<<(xTot + 255) / 256, 256, 0, stream>>>(hist, hs0, h, zr, X, Xsw, 0, t, Cin, Cip, 1);
      ag_graph_gemm_kernel<<<dim3(Mcols / 64, N_ / 128), 128, 0, stream>>>(A, Xsw, Y, Mcols);
      ag_node_upd_kernel<<<N_, 128, 0, stream>>>(X, Y, PWu0, PBu0, zr, h, hs0, Cip, t, 0);
    }
  }

  // 4. layer 1: Cin=64, Cip=128, Mcols = 8192
  ag_zero_kernel<<<(hElems + 255) / 256, 256, 0, stream>>>(h, hElems);
  {
    const int Cip = 128, Cin = 64, Mcols = B_ * Cip;
    const int xTot = N_ * B_ * Cip;
    for (int t = 0; t < L_; ++t) {
      ag_build_x_kernel<<<(xTot + 255) / 256, 256, 0, stream>>>(hist, hs0, h, zr, X, Xsw, 1, t, Cin, Cip, 0);
      ag_graph_gemm_kernel<<<dim3(Mcols / 64, N_ / 128), 128, 0, stream>>>(A, Xsw, Y, Mcols);
      ag_node_gate_kernel<<<N_, 256, 0, stream>>>(X, Y, PWg1, PBg1, zr, Cip);
      ag_build_x_kernel<<<(xTot + 255) / 256, 256, 0, stream>>>(hist, hs0, h, zr, X, Xsw, 1, t, Cin, Cip, 1);
      ag_graph_gemm_kernel<<<dim3(Mcols / 64, N_ / 128), 128, 0, stream>>>(A, Xsw, Y, Mcols);
      ag_node_upd_kernel<<<N_, 128, 0, stream>>>(X, Y, PWu1, PBu1, zr, h, hs0, Cip, t, 1);
    }
  }

  // 5. output head
  {
    int tot = B_ * HOR_ * N_;
    ag_out_kernel<<<(tot + 255) / 256, 256, 0, stream>>>(h, cw, cb, out);
  }
}
